// ComplexS4ModelLayer_48137993453868
// MI455X (gfx1250) — matrix operand order repaired and checked by reading, not compiled, not run on hardware
//
#include <hip/hip_runtime.h>
#include <hip/hip_bf16.h>
#include <math.h>

typedef __attribute__((ext_vector_type(16))) _Float16     v16h;
typedef __attribute__((ext_vector_type(8)))  _Float16     v8h;
typedef __attribute__((ext_vector_type(8)))  float        v8f;
typedef __attribute__((ext_vector_type(8)))  unsigned int v8u;

static constexpr int B_  = 8;
static constexpr int T_  = 2048;
static constexpr int DM_ = 256;
static constexpr int NL_ = 4;
static constexpr int NS_ = 32;    // complex state modes
static constexpr int DIN_ = 514;
static constexpr int KENC_ = 544; // 514 padded to mult of 32
static constexpr int GDEC_ = 544; // 514 padded to mult of 32 (N tile = 32)

// ---------------- fragment loaders (CDNA5 16-bit WMMA layouts, wave32) -------
// A 16x32 f16: lanes 0-15 hold M=0..15 (K groups {0..7,16..23}); lanes 16-31 same M, K+8.
__device__ __forceinline__ v16h load_a_frag(const _Float16* base, int ld, int row0, int k0, int lane) {
    int m  = lane & 15;
    int g2 = lane >> 4;
    const _Float16* p = base + (size_t)(row0 + m) * ld + k0 + g2 * 8;
    union { v16h v; v8h h[2]; } f;
    f.h[0] = *(const v8h*)p;
    f.h[1] = *(const v8h*)(p + 16);
    return f.v;
}
// B 32x16 f16: lane n = lane&15 (column), K groups {0..7,16..23}; lanes 16-31 K+8.
__device__ __forceinline__ v16h load_b_frag(const _Float16* base, int ld, int g0, int k0, int lane) {
    int n  = lane & 15;
    int g2 = lane >> 4;
    const _Float16* p = base + (size_t)(g0 + n) * ld + k0 + g2 * 8;
    union { v16h v; v8h h[2]; } f;
    f.h[0] = *(const v8h*)p;
    f.h[1] = *(const v8h*)(p + 16);
    return f.v;
}
// negate packed f16 fragment by flipping sign bits (cheaper than extra loads)
__device__ __forceinline__ v16h neg_frag(v16h a) {
    union { v16h h; v8u u; } f;
    f.h = a;
    f.u = f.u ^ 0x80008000u;
    return f.h;
}

#define WMMA_F16(A, B, C) __builtin_amdgcn_wmma_f32_16x16x32_f16(false, (A), false, (B), (short)0, (C), false, false)

// ---------------- complex GEMM, 32x32 output tile per wave (2x2 fragments) ---
// Y^T(t,g): Yr = Xr*Wr - Xi*Wi ; Yi = Xi*Wr + Xr*Wi
// One wave per block; full register file (no occupancy cap) to avoid spills.
__global__ void __launch_bounds__(32, 1)
cgemm_kernel(const _Float16* __restrict__ XrT, const _Float16* __restrict__ XiT,
             int ldx, size_t xBatch,
             const _Float16* __restrict__ Wr, const _Float16* __restrict__ Wi, int ldw,
             const float* __restrict__ br, const float* __restrict__ bi,
             int G, int K,
             float* __restrict__ Yr, float* __restrict__ Yi,
             size_t yBatch, int ldy, int es) {
    int lane = threadIdx.x;
    int g0 = blockIdx.x * 32;
    int t0 = blockIdx.y * 32;
    int b  = blockIdx.z;
    const _Float16* xr = XrT + (size_t)b * xBatch;
    const _Float16* xi = XiT + (size_t)b * xBatch;
    v8f aR00 = {}, aR01 = {}, aR10 = {}, aR11 = {};
    v8f aI00 = {}, aI01 = {}, aI10 = {}, aI11 = {};
    for (int k0 = 0; k0 < K; k0 += 32) {
        v16h ar0 = load_a_frag(xr, ldx, t0,      k0, lane);
        v16h ar1 = load_a_frag(xr, ldx, t0 + 16, k0, lane);
        v16h ai0 = load_a_frag(xi, ldx, t0,      k0, lane);
        v16h ai1 = load_a_frag(xi, ldx, t0 + 16, k0, lane);
        v16h bw0 = load_b_frag(Wr, ldw, g0,      k0, lane);
        v16h bw1 = load_b_frag(Wr, ldw, g0 + 16, k0, lane);
        v16h bi0 = load_b_frag(Wi, ldw, g0,      k0, lane);
        v16h bi1 = load_b_frag(Wi, ldw, g0 + 16, k0, lane);
        v16h bn0 = neg_frag(bi0);
        v16h bn1 = neg_frag(bi1);
        // first contribution to each of the 8 independent accumulators ...
        aR00 = WMMA_F16(ar0, bw0, aR00);
        aR01 = WMMA_F16(ar0, bw1, aR01);
        aR10 = WMMA_F16(ar1, bw0, aR10);
        aR11 = WMMA_F16(ar1, bw1, aR11);
        aI00 = WMMA_F16(ai0, bw0, aI00);
        aI01 = WMMA_F16(ai0, bw1, aI01);
        aI10 = WMMA_F16(ai1, bw0, aI10);
        aI11 = WMMA_F16(ai1, bw1, aI11);
        // ... then the second contribution (C-chains are well separated)
        aR00 = WMMA_F16(ai0, bn0, aR00);
        aR01 = WMMA_F16(ai0, bn1, aR01);
        aR10 = WMMA_F16(ai1, bn0, aR10);
        aR11 = WMMA_F16(ai1, bn1, aR11);
        aI00 = WMMA_F16(ar0, bi0, aI00);
        aI01 = WMMA_F16(ar0, bi1, aI01);
        aI10 = WMMA_F16(ar1, bi0, aI10);
        aI11 = WMMA_F16(ar1, bi1, aI11);
    }
    float* yr = Yr + (size_t)b * yBatch;
    float* yi = Yi + (size_t)b * yBatch;
    v8f* accR[2][2] = { { &aR00, &aR01 }, { &aR10, &aR11 } };
    v8f* accI[2][2] = { { &aI00, &aI01 }, { &aI10, &aI11 } };
    #pragma unroll
    for (int nTile = 0; nTile < 2; ++nTile) {
        int g = g0 + nTile * 16 + (lane & 15);
        int gc = g < G ? g : (G - 1);
        float biasR = br[gc], biasI = bi[gc];
        if (g < G) {
            #pragma unroll
            for (int mTile = 0; mTile < 2; ++mTile) {
                v8f vR = *accR[mTile][nTile];
                v8f vI = *accI[mTile][nTile];
                #pragma unroll
                for (int r = 0; r < 8; ++r) {
                    int m = t0 + mTile * 16 + r + 8 * (lane >> 4);
                    size_t off = ((size_t)g * ldy + m) * (size_t)es;
                    yr[off] = vR[r] + biasR;
                    yi[off] = vI[r] + biasI;
                }
            }
        }
    }
}

// ---------------- x deinterleave + transpose + pad: (B,514,T,2)f32 -> (B,T,544)f16 x2
__global__ void deint_x_kernel(const float* __restrict__ x,
                               _Float16* __restrict__ XrT, _Float16* __restrict__ XiT) {
    __shared__ float sr[32][33];
    __shared__ float si[32][33];
    int tx = threadIdx.x, ty = threadIdx.y;
    int t0 = blockIdx.x * 32, d0 = blockIdx.y * 32, b = blockIdx.z;
    #pragma unroll
    for (int j = 0; j < 4; ++j) {
        int d = d0 + ty + 8 * j;
        float vr = 0.f, vi = 0.f;
        if (d < DIN_) {
            const float* p = x + (((size_t)b * DIN_ + d) * T_ + (t0 + tx)) * 2;
            vr = p[0]; vi = p[1];
        }
        sr[ty + 8 * j][tx] = vr;
        si[ty + 8 * j][tx] = vi;
    }
    __syncthreads();
    #pragma unroll
    for (int j = 0; j < 4; ++j) {
        int t = t0 + ty + 8 * j;
        int d = d0 + tx;
        size_t off = ((size_t)b * T_ + t) * KENC_ + d;
        XrT[off] = (_Float16)sr[tx][ty + 8 * j];
        XiT[off] = (_Float16)si[tx][ty + 8 * j];
    }
}

// ---------------- weight f32 -> f16, zero padded ------------------------------
__global__ void cvt_w_kernel(const float* __restrict__ src, int G, int H,
                             _Float16* __restrict__ dst, int Gp, int Hp) {
    int idx = blockIdx.x * blockDim.x + threadIdx.x;
    if (idx >= Gp * Hp) return;
    int g = idx / Hp, h = idx % Hp;
    float v = (g < G && h < H) ? src[(size_t)g * H + h] : 0.f;
    dst[idx] = (_Float16)v;
}

// ---------------- S4D diagonal recurrence + skip + GELU ----------------------
// one wave per (b,h); lane = complex mode n. Exact equivalent of the FFT conv.
__global__ void s4d_rec_kernel(const float* __restrict__ zr, const float* __restrict__ zi,
                               const float* __restrict__ log_dt, const float* __restrict__ log_A_real,
                               const float* __restrict__ A_imag, const float* __restrict__ C_r,
                               const float* __restrict__ C_i, const float* __restrict__ Dp,
                               _Float16* __restrict__ yhr, _Float16* __restrict__ yhi) {
    int lane = threadIdx.x;
    int h = blockIdx.x % DM_;
    int b = blockIdx.x / DM_;
    float dt = __expf(log_dt[h]);
    int hn = h * NS_ + lane;
    float Ar = -__expf(log_A_real[hn]);
    float Ai = A_imag[hn];
    float ee = __expf(Ar * dt);
    float wr = ee * __cosf(Ai * dt);
    float wi = ee * __sinf(Ai * dt);
    // Ct = (Cr + i Ci) * (exp(dtA) - 1) / A
    float m1r = wr - 1.f, m1i = wi;
    float den = Ar * Ar + Ai * Ai;
    float qr = (m1r * Ar + m1i * Ai) / den;
    float qi = (m1i * Ar - m1r * Ai) / den;
    float cr = C_r[hn], ci = C_i[hn];
    float Ctr = cr * qr - ci * qi;
    float Cti = cr * qi + ci * qr;
    float Dh = Dp[h];
    const float* ur = zr + ((size_t)b * DM_ + h) * T_;
    const float* ui = zi + ((size_t)b * DM_ + h) * T_;
    _Float16* outr = yhr + ((size_t)b * DM_ + h) * T_;
    _Float16* outi = yhi + ((size_t)b * DM_ + h) * T_;
    float sr = 0.f, si = 0.f;
    for (int t0 = 0; t0 < T_; t0 += 32) {
        float uvr = ur[t0 + lane];
        float uvi = ui[t0 + lane];
        float or_ = 0.f, oi_ = 0.f;
        for (int j = 0; j < 32; ++j) {
            float xr_ = __shfl(uvr, j, 32);
            float xi_ = __shfl(uvi, j, 32);
            float nsr = fmaf(wr, sr, fmaf(-wi, si, xr_));
            float nsi = fmaf(wr, si, fmaf( wi, sr, xi_));
            sr = nsr; si = nsi;
            float pr = Ctr * sr - Cti * si;
            float pi = Ctr * si + Cti * sr;
            #pragma unroll
            for (int off = 16; off > 0; off >>= 1) {
                pr += __shfl_xor(pr, off, 32);
                pi += __shfl_xor(pi, off, 32);
            }
            pr = fmaf(Dh, xr_, pr);
            pi = fmaf(Dh, xi_, pi);
            float gr = 0.5f * pr * (1.f + tanhf(0.7978845608f * (pr + 0.044715f * pr * pr * pr)));
            float gi = 0.5f * pi * (1.f + tanhf(0.7978845608f * (pi + 0.044715f * pi * pi * pi)));
            if (j == lane) { or_ = gr; oi_ = gi; }
        }
        outr[t0 + lane] = (_Float16)or_;
        outi[t0 + lane] = (_Float16)oi_;
    }
}

// ---------------- f16 transpose (B,256,T) -> (B,T,256) -----------------------
__global__ void tr16_kernel(const _Float16* __restrict__ in, _Float16* __restrict__ out) {
    __shared__ _Float16 s[32][33];
    int tx = threadIdx.x, ty = threadIdx.y;
    int t0 = blockIdx.x * 32, h0 = blockIdx.y * 32, b = blockIdx.z;
    #pragma unroll
    for (int j = 0; j < 4; ++j)
        s[ty + 8 * j][tx] = in[((size_t)b * DM_ + h0 + ty + 8 * j) * T_ + t0 + tx];
    __syncthreads();
    #pragma unroll
    for (int j = 0; j < 4; ++j)
        out[((size_t)b * T_ + t0 + ty + 8 * j) * DM_ + h0 + tx] = s[tx][ty + 8 * j];
}

// ---------------- residual add + channel LayerNorm (in-place safe) -----------
__global__ void ln_kernel(const float* __restrict__ z, const float* __restrict__ o,
                          const float* __restrict__ gamma, const float* __restrict__ beta,
                          float* __restrict__ zout, _Float16* __restrict__ zh) {
    __shared__ float ssum[8][32];
    __shared__ float ssq[8][32];
    __shared__ float smv[2][32];
    int tx = threadIdx.x, ty = threadIdx.y;
    int t = blockIdx.x * 32 + tx;
    int b = blockIdx.y;
    size_t base = (size_t)b * DM_ * T_ + t;
    float v[32];
    float sum = 0.f, sq = 0.f;
    #pragma unroll
    for (int k = 0; k < 32; ++k) {
        int c = ty + 8 * k;
        float val = z[base + (size_t)c * T_] + o[base + (size_t)c * T_];
        v[k] = val; sum += val; sq = fmaf(val, val, sq);
    }
    ssum[ty][tx] = sum; ssq[ty][tx] = sq;
    __syncthreads();
    if (ty == 0) {
        float s = 0.f, q = 0.f;
        #pragma unroll
        for (int k = 0; k < 8; ++k) { s += ssum[k][tx]; q += ssq[k][tx]; }
        float mean = s * (1.f / DM_);
        float var = q * (1.f / DM_) - mean * mean;
        smv[0][tx] = mean;
        smv[1][tx] = rsqrtf(var + 1e-5f);
    }
    __syncthreads();
    float mean = smv[0][tx], rstd = smv[1][tx];
    #pragma unroll
    for (int k = 0; k < 32; ++k) {
        int c = ty + 8 * k;
        float val = (v[k] - mean) * rstd * gamma[c] + beta[c];
        zout[base + (size_t)c * T_] = val;
        zh[base + (size_t)c * T_] = (_Float16)val;
    }
}

extern "C" void kernel_launch(void* const* d_in, const int* in_sizes, int n_in,
                              void* d_out, int out_size, void* d_ws, size_t ws_size,
                              hipStream_t stream) {
    const float* x        = (const float*)d_in[0];
    const float* enc_Wr   = (const float*)d_in[1];
    const float* enc_Wi   = (const float*)d_in[2];
    const float* enc_br   = (const float*)d_in[3];
    const float* enc_bi   = (const float*)d_in[4];
    const float* log_dt   = (const float*)d_in[5];
    const float* log_A    = (const float*)d_in[6];
    const float* A_imag   = (const float*)d_in[7];
    const float* C_r      = (const float*)d_in[8];
    const float* C_i      = (const float*)d_in[9];
    const float* Dp       = (const float*)d_in[10];
    const float* out_Wr   = (const float*)d_in[11];
    const float* out_Wi   = (const float*)d_in[12];
    const float* out_br   = (const float*)d_in[13];
    const float* out_bi   = (const float*)d_in[14];
    const float* ln_gr    = (const float*)d_in[15];
    const float* ln_gi    = (const float*)d_in[16];
    const float* ln_br    = (const float*)d_in[17];
    const float* ln_bi    = (const float*)d_in[18];
    const float* dec_Wr   = (const float*)d_in[19];
    const float* dec_Wi   = (const float*)d_in[20];
    const float* dec_br   = (const float*)d_in[21];
    const float* dec_bi   = (const float*)d_in[22];
    float* out = (float*)d_out;

    char* ws = (char*)d_ws;
    size_t cur = 0;
    auto alloc = [&](size_t bytes) -> char* {
        char* p = ws + cur;
        cur += (bytes + 255) & ~(size_t)255;
        return p;
    };
    const size_t BHT = (size_t)B_ * DM_ * T_;
    _Float16* xrT    = (_Float16*)alloc((size_t)B_ * T_ * KENC_ * 2);
    _Float16* xiT    = (_Float16*)alloc((size_t)B_ * T_ * KENC_ * 2);
    _Float16* encWrH = (_Float16*)alloc((size_t)DM_ * KENC_ * 2);
    _Float16* encWiH = (_Float16*)alloc((size_t)DM_ * KENC_ * 2);
    _Float16* outWrH = (_Float16*)alloc((size_t)NL_ * DM_ * DM_ * 2);
    _Float16* outWiH = (_Float16*)alloc((size_t)NL_ * DM_ * DM_ * 2);
    _Float16* decWrH = (_Float16*)alloc((size_t)GDEC_ * DM_ * 2);
    _Float16* decWiH = (_Float16*)alloc((size_t)GDEC_ * DM_ * 2);
    float*    zr     = (float*)   alloc(BHT * 4);
    float*    zi     = (float*)   alloc(BHT * 4);
    _Float16* zhr    = (_Float16*)alloc(BHT * 2);
    _Float16* zhi    = (_Float16*)alloc(BHT * 2);
    _Float16* yhr    = (_Float16*)alloc(BHT * 2);
    _Float16* yhi    = (_Float16*)alloc(BHT * 2);
    _Float16* yTr    = (_Float16*)alloc(BHT * 2);
    _Float16* yTi    = (_Float16*)alloc(BHT * 2);
    _Float16* zTr    = (_Float16*)alloc(BHT * 2);
    _Float16* zTi    = (_Float16*)alloc(BHT * 2);
    float*    orr    = (float*)   alloc(BHT * 4);
    float*    oii    = (float*)   alloc(BHT * 4);
    (void)ws_size; (void)n_in; (void)in_sizes; (void)out_size;

    // --- weight conversion ---
    cvt_w_kernel<<<(DM_ * KENC_ + 255) / 256, 256, 0, stream>>>(enc_Wr, DM_, DIN_, encWrH, DM_, KENC_);
    cvt_w_kernel<<<(DM_ * KENC_ + 255) / 256, 256, 0, stream>>>(enc_Wi, DM_, DIN_, encWiH, DM_, KENC_);
    cvt_w_kernel<<<(NL_ * DM_ * DM_ + 255) / 256, 256, 0, stream>>>(out_Wr, NL_ * DM_, DM_, outWrH, NL_ * DM_, DM_);
    cvt_w_kernel<<<(NL_ * DM_ * DM_ + 255) / 256, 256, 0, stream>>>(out_Wi, NL_ * DM_, DM_, outWiH, NL_ * DM_, DM_);
    cvt_w_kernel<<<(GDEC_ * DM_ + 255) / 256, 256, 0, stream>>>(dec_Wr, DIN_, DM_, decWrH, GDEC_, DM_);
    cvt_w_kernel<<<(GDEC_ * DM_ + 255) / 256, 256, 0, stream>>>(dec_Wi, DIN_, DM_, decWiH, GDEC_, DM_);

    // --- input deinterleave/transpose ---
    deint_x_kernel<<<dim3(T_ / 32, KENC_ / 32, B_), dim3(32, 8), 0, stream>>>(x, xrT, xiT);

    // --- encoder complex GEMM -> z (B,256,T) ---
    cgemm_kernel<<<dim3(DM_ / 32, T_ / 32, B_), 32, 0, stream>>>(
        xrT, xiT, KENC_, (size_t)T_ * KENC_,
        encWrH, encWiH, KENC_,
        enc_br, enc_bi, DM_, KENC_,
        zr, zi, (size_t)DM_ * T_, T_, 1);

    // --- layers ---
    for (int l = 0; l < NL_; ++l) {
        s4d_rec_kernel<<<B_ * DM_, 32, 0, stream>>>(
            zr, zi,
            log_dt + l * DM_, log_A + (size_t)l * DM_ * NS_, A_imag + (size_t)l * DM_ * NS_,
            C_r + (size_t)l * DM_ * NS_, C_i + (size_t)l * DM_ * NS_, Dp + l * DM_,
            yhr, yhi);
        tr16_kernel<<<dim3(T_ / 32, DM_ / 32, B_), dim3(32, 8), 0, stream>>>(yhr, yTr);
        tr16_kernel<<<dim3(T_ / 32, DM_ / 32, B_), dim3(32, 8), 0, stream>>>(yhi, yTi);
        cgemm_kernel<<<dim3(DM_ / 32, T_ / 32, B_), 32, 0, stream>>>(
            yTr, yTi, DM_, (size_t)T_ * DM_,
            outWrH + (size_t)l * DM_ * DM_, outWiH + (size_t)l * DM_ * DM_, DM_,
            out_br + l * DM_, out_bi + l * DM_, DM_, DM_,
            orr, oii, (size_t)DM_ * T_, T_, 1);
        ln_kernel<<<dim3(T_ / 32, B_), dim3(32, 8), 0, stream>>>(zr, orr, ln_gr + l * DM_, ln_br + l * DM_, zr, zhr);
        ln_kernel<<<dim3(T_ / 32, B_), dim3(32, 8), 0, stream>>>(zi, oii, ln_gi + l * DM_, ln_bi + l * DM_, zi, zhi);
    }

    // --- decoder ---
    tr16_kernel<<<dim3(T_ / 32, DM_ / 32, B_), dim3(32, 8), 0, stream>>>(zhr, zTr);
    tr16_kernel<<<dim3(T_ / 32, DM_ / 32, B_), dim3(32, 8), 0, stream>>>(zhi, zTi);
    cgemm_kernel<<<dim3(GDEC_ / 32, T_ / 32, B_), 32, 0, stream>>>(
        zTr, zTi, DM_, (size_t)T_ * DM_,
        decWrH, decWiH, DM_,
        dec_br, dec_bi, DIN_, DM_,
        out, out + 1, (size_t)DIN_ * T_ * 2, T_, 2);
}
